// Attention_22067541966988
// MI455X (gfx1250) — matrix operand order repaired and checked by reading, not compiled, not run on hardware
//
#include <hip/hip_runtime.h>
#include <hip/hip_bf16.h>
#include <math.h>

// Shapes (compile-time constants from the reference)
#define Bb 2
#define Nn 1024
#define Cc 256
#define Pp 64
#define Hh 8
#define Dd 32
#define TDd 256

typedef __attribute__((ext_vector_type(16))) _Float16 v16h;
typedef __attribute__((ext_vector_type(8)))  _Float16 v8h;
typedef __attribute__((ext_vector_type(8)))  float    v8f;

// ---------------------------------------------------------------------------
// WMMA fragment loaders (CDNA5 16x16x32 f16, wave32) — k = 16*hi + s in both A and B
// ---------------------------------------------------------------------------

// A fragment (16x32, row-major f32 source, convert to f16). lane&15 = row.
// Per lane the A slice is one contiguous 64B run: k = hi*16 + [0..15]
//  -> 4 aligned float4 loads.
__device__ __forceinline__ v16h load_a_f32(const float* __restrict__ base,
                                           int stride, int lane) {
  const int r = lane & 15, hi = lane >> 4;
  const float* row = base + (long)r * stride + hi * 16;
  const float4 x0 = *(const float4*)(row);
  const float4 x1 = *(const float4*)(row + 4);
  const float4 x2 = *(const float4*)(row + 8);
  const float4 x3 = *(const float4*)(row + 12);
  v16h a;
  a[0]  = (_Float16)x0.x;  a[1]  = (_Float16)x0.y;
  a[2]  = (_Float16)x0.z;  a[3]  = (_Float16)x0.w;
  a[4]  = (_Float16)x1.x;  a[5]  = (_Float16)x1.y;
  a[6]  = (_Float16)x1.z;  a[7]  = (_Float16)x1.w;
  a[8]  = (_Float16)x2.x;  a[9]  = (_Float16)x2.y;
  a[10] = (_Float16)x2.z;  a[11] = (_Float16)x2.w;
  a[12] = (_Float16)x3.x;  a[13] = (_Float16)x3.y;
  a[14] = (_Float16)x3.z;  a[15] = (_Float16)x3.w;
  return a;
}

// A fragment from f16 row-major (works for global or LDS pointers).
__device__ __forceinline__ v16h load_a_f16(const _Float16* __restrict__ base,
                                           int stride, int lane) {
  const int r = lane & 15, hi = lane >> 4;
  const _Float16* row = base + r * stride;
  v16h a;
#pragma unroll
  for (int i = 0; i < 8; ++i) {
    const int k0 = hi * 16 + 2 * i;
    a[2 * i]     = row[k0];
    a[2 * i + 1] = row[k0 + 1];
  }
  return a;
}

// B fragment (32x16) from f32 [K x Nout] row-major weight W. col = n0 + lane&15,
// lanes 0-15 hold K=0..15, lanes 16-31 hold K=16..31 (2 per VGPR).
__device__ __forceinline__ v16h load_b_f32(const float* __restrict__ W,
                                           int ldw, int n0, int lane) {
  const int c = n0 + (lane & 15), hi = lane >> 4;
  v16h b;
#pragma unroll
  for (int i = 0; i < 8; ++i) {
    const int k = hi * 16 + 2 * i;
    b[2 * i]     = (_Float16)W[(long)k * ldw + c];
    b[2 * i + 1] = (_Float16)W[(long)(k + 1) * ldw + c];
  }
  return b;
}

// B fragment for S = Q*K^T : B[k=d][n=key], K tile stored [key][d] f16 (pairs
// along d contiguous).
__device__ __forceinline__ v16h load_b_kT(const _Float16* __restrict__ kb,
                                          int key0, int lane) {
  const int key = key0 + (lane & 15), hi = lane >> 4;
  const _Float16* p = kb + (long)key * Dd;
  v16h b;
#pragma unroll
  for (int i = 0; i < 8; ++i) {
    const int d = hi * 16 + 2 * i;
    b[2 * i]     = p[d];
    b[2 * i + 1] = p[d + 1];
  }
  return b;
}

// B fragment for O += P*V : B[k=key][n=d], V stored transposed [d][key] f16
// (pairs along key contiguous).
__device__ __forceinline__ v16h load_b_vT(const _Float16* __restrict__ vt,
                                          int d0, int m0, int lane) {
  const int d = d0 + (lane & 15), hi = lane >> 4;
  const _Float16* p = vt + (long)d * Nn + m0 + hi * 16;
  v16h b;
#pragma unroll
  for (int i = 0; i < 8; ++i) {
    b[2 * i]     = p[2 * i];
    b[2 * i + 1] = p[2 * i + 1];
  }
  return b;
}

__device__ __forceinline__ v8f wmma_f16(v16h a, v16h b, v8f c) {
  return __builtin_amdgcn_wmma_f32_16x16x32_f16(false, a, false, b,
                                                (short)0, c, false, false);
}

// ---------------------------------------------------------------------------
// Kernel 1: bias[b,h,n,m] = pair[b,n,m,:]@Wb[:,h] + bb[h]   (f16 out, 32 MB)
// WMMA GEMM: rows = flattened (b,n,m), K = P = 64, cols = heads (8, padded 16).
// One wave processes 8 consecutive 16-row tiles (128 rows); Wb fragments and
// the bb accumulator seed are hoisted. Streams the 512 MB pair tensor once.
// ---------------------------------------------------------------------------
__global__ void bias_kernel(const float* __restrict__ pair,
                            const float* __restrict__ Wb,
                            const float* __restrict__ bbv,
                            _Float16* __restrict__ bias) {
  const int lane = threadIdx.x & 31;
  const int wave = threadIdx.x >> 5;
  const int lane16 = lane & 15, hi = lane >> 4;

  // B fragments for Wb [64 x 16pad]: B[k=p][n=h] = Wb[p*8+h], h>=8 -> 0
  v16h b1, b2;
#pragma unroll
  for (int i = 0; i < 8; ++i) {
    const int k = hi * 16 + 2 * i;
    b1[2 * i]     = (_Float16)((lane16 < Hh) ? Wb[(k)      * Hh + lane16] : 0.0f);
    b1[2 * i + 1] = (_Float16)((lane16 < Hh) ? Wb[(k + 1)  * Hh + lane16] : 0.0f);
    b2[2 * i]     = (_Float16)((lane16 < Hh) ? Wb[(k + 32) * Hh + lane16] : 0.0f);
    b2[2 * i + 1] = (_Float16)((lane16 < Hh) ? Wb[(k + 33) * Hh + lane16] : 0.0f);
  }
  const float bbc = (lane16 < Hh) ? bbv[lane16] : 0.0f;

  const long wtile  = (long)blockIdx.x * 4 + wave;  // 0 .. 16383
  const long nmBase = wtile << 7;                   // 128 rows per wave

#pragma unroll
  for (int t = 0; t < 8; ++t) {
    const long nm0 = nmBase + t * 16;
    const float* abase = pair + nm0 * Pp;

    v8f acc;
#pragma unroll
    for (int j = 0; j < 8; ++j) acc[j] = bbc;

    v16h a1 = load_a_f32(abase, Pp, lane);       // p = 0..31
    v16h a2 = load_a_f32(abase + 32, Pp, lane);  // p = 32..63
    acc = wmma_f16(a1, b1, acc);
    acc = wmma_f16(a2, b2, acc);

    // lane holds rows nm0 + 8*hi + j (j=0..7), fixed col h = lane16:
    // pack 8 f16 and store 16B at bias[((b*H+h)<<20) + nm_local]
    if (lane16 < Hh) {
      const long r0   = nm0 + 8 * hi;
      const long bidx = r0 >> 20;         // N*N = 2^20
      const long nml  = r0 & 0xFFFFF;
      v8h ov;
#pragma unroll
      for (int j = 0; j < 8; ++j) ov[j] = (_Float16)acc[j];
      *(v8h*)(bias + ((bidx * Hh + lane16) << 20) + nml) = ov;
    }
  }
}

// ---------------------------------------------------------------------------
// Kernel 2: projections  out = X[2048,256] @ W[256,256]  (WMMA f16, f32 accum)
// mode 0: Q -> q_p[B,H,N,D] f16, scaled by D^-1/2
// mode 1: K -> k_p[B,H,N,D] f16
// mode 2: V -> vT [B,H,D,N] f16 (transposed)
// mode 3: gate = sigmoid(X@Wg + bg) -> f32 [B*N, 256]
// ---------------------------------------------------------------------------
__global__ void proj_kernel(const float* __restrict__ X,
                            const float* __restrict__ W,
                            const float* __restrict__ bvec,
                            _Float16* __restrict__ outH,
                            float* __restrict__ outG, int mode) {
  const int lane = threadIdx.x & 31;
  const int wave = threadIdx.x >> 5;
  const int m0 = blockIdx.x * 16;              // row tile in [0,2048)
  const int n0 = blockIdx.y * 64 + wave * 16;  // col tile in [0,256)

  v8f acc = {};
#pragma unroll
  for (int k0 = 0; k0 < Cc; k0 += 32) {
    v16h a = load_a_f32(X + (long)m0 * Cc + k0, Cc, lane);
    v16h b = load_b_f32(W + (long)k0 * TDd, TDd, n0, lane);
    acc = wmma_f16(a, b, acc);
  }

  const int hi = lane >> 4;
  const int c  = n0 + (lane & 15);
  const float qscale = 0.17677669529663687f;  // 32^-0.5
#pragma unroll
  for (int j = 0; j < 8; ++j) {
    const int r    = m0 + j + 8 * hi;  // global row (b*N + nseq)
    const int bidx = r >> 10;
    const int nseq = r & 1023;
    const float v  = acc[j];
    if (mode == 0) {
      const int h = c >> 5, d = c & 31;
      outH[((long)((bidx * Hh + h) << 10) + nseq) * Dd + d] =
          (_Float16)(v * qscale);
    } else if (mode == 1) {
      const int h = c >> 5, d = c & 31;
      outH[((long)((bidx * Hh + h) << 10) + nseq) * Dd + d] = (_Float16)v;
    } else if (mode == 2) {
      const int h = c >> 5, d = c & 31;
      outH[((long)((bidx * Hh + h) * Dd + d) << 10) + nseq] = (_Float16)v;
    } else {
      const float g = v + bvec[c];
      outG[(long)r * TDd + c] = 1.0f / (1.0f + __expf(-g));
    }
  }
}

// ---------------------------------------------------------------------------
// Kernel 3: fused flash attention. One wave per (b, h, 16-query tile).
// ---------------------------------------------------------------------------
__global__ void attn_kernel(const _Float16* __restrict__ qp,
                            const _Float16* __restrict__ kp,
                            const _Float16* __restrict__ vt,
                            const _Float16* __restrict__ bias,
                            const float* __restrict__ mask,
                            const float* __restrict__ gate,
                            _Float16* __restrict__ att) {
  __shared__ _Float16 lds[4][16 * 32];
  const int lane = threadIdx.x & 31;
  const int wave = threadIdx.x >> 5;
  const int tile = blockIdx.x * 4 + wave;  // 0..1023
  const int q0 = (tile & 63) << 4;         // 64 q-tiles per (b,h)
  const int bh = tile >> 6;                // 0..15
  const int b  = bh >> 3;
  const int h  = bh & 7;

  const _Float16* qb = qp + ((long)bh << 10) * Dd;    // [N,D]
  const _Float16* kb = kp + ((long)bh << 10) * Dd;    // [N,D]
  const _Float16* vb = vt + ((long)bh << 10) * Dd;    // [D,N]
  const _Float16* bi = bias + ((long)bh << 20);       // [N,N]
  const float*    mk = mask + ((long)b << 10);        // [N]
  _Float16* myl = lds[wave];

  const int lane16 = lane & 15, hi = lane >> 4;
  const v16h qa = load_a_f16(qb + (long)q0 * Dd, Dd, lane);

  v8f o1 = {}, o2 = {};
  float mrow[8], lrow[8];
#pragma unroll
  for (int j = 0; j < 8; ++j) { mrow[j] = -1e30f; lrow[j] = 0.0f; }

  for (int m0 = 0; m0 < Nn; m0 += 32) {
    v16h bk1 = load_b_kT(kb, m0, lane);
    v16h bk2 = load_b_kT(kb, m0 + 16, lane);
    v8f s1 = {}, s2 = {};
    s1 = wmma_f16(qa, bk1, s1);
    s2 = wmma_f16(qa, bk2, s2);

    const float mk1 = mk[m0 + lane16];
    const float mk2 = mk[m0 + 16 + lane16];

#pragma unroll
    for (int j = 0; j < 8; ++j) {
      const int q = q0 + j + 8 * hi;
      const _Float16* bp = bi + (long)q * Nn + m0;
      float a1 = s1[j] + (float)bp[lane16] + mk1;
      float a2 = s2[j] + (float)bp[16 + lane16] + mk2;

      // row max across the 16 lanes of this half-wave
      float mx = fmaxf(a1, a2);
      mx = fmaxf(mx, __shfl_xor(mx, 1));
      mx = fmaxf(mx, __shfl_xor(mx, 2));
      mx = fmaxf(mx, __shfl_xor(mx, 4));
      mx = fmaxf(mx, __shfl_xor(mx, 8));
      const float mnew = fmaxf(mrow[j], mx);
      const float sc   = __expf(mrow[j] - mnew);
      const float e1   = __expf(a1 - mnew);
      const float e2   = __expf(a2 - mnew);
      float rs = e1 + e2;
      rs += __shfl_xor(rs, 1);
      rs += __shfl_xor(rs, 2);
      rs += __shfl_xor(rs, 4);
      rs += __shfl_xor(rs, 8);
      lrow[j] = lrow[j] * sc + rs;
      mrow[j] = mnew;
      o1[j] *= sc;
      o2[j] *= sc;

      // stage P tile (C-layout -> LDS, 16 rows x 32 keys)
      const int row = j + 8 * hi;
      myl[row * 32 + lane16]      = (_Float16)e1;
      myl[row * 32 + 16 + lane16] = (_Float16)e2;
    }
    // same-wave LDS store->load hazard: wait for DS writes to land
    asm volatile("s_wait_dscnt 0x0" ::: "memory");

    v16h pa  = load_a_f16(myl, 32, lane);       // A-layout P (16x32)
    v16h bv1 = load_b_vT(vb, 0, m0, lane);
    v16h bv2 = load_b_vT(vb, 16, m0, lane);
    o1 = wmma_f16(pa, bv1, o1);
    o2 = wmma_f16(pa, bv2, o2);
  }

  // epilogue: normalize, gate, store f16 att[B*N, 256]
#pragma unroll
  for (int j = 0; j < 8; ++j) {
    const int q = q0 + j + 8 * hi;
    const float inv = (lrow[j] > 0.0f) ? 1.0f / lrow[j] : 0.0f;
    const long rowoff = (((long)b << 10) + q) * TDd + h * Dd;
    const float g1 = gate[rowoff + lane16];
    const float g2 = gate[rowoff + 16 + lane16];
    att[rowoff + lane16]      = (_Float16)(o1[j] * inv * g1);
    att[rowoff + 16 + lane16] = (_Float16)(o2[j] * inv * g2);
  }
}

// ---------------------------------------------------------------------------
// Kernel 4: out = att(f16)[2048,256] @ Wo[256,256] + bo  -> f32
// ---------------------------------------------------------------------------
__global__ void out_gemm(const _Float16* __restrict__ A,
                         const float* __restrict__ W,
                         const float* __restrict__ bo,
                         float* __restrict__ out) {
  const int lane = threadIdx.x & 31;
  const int wave = threadIdx.x >> 5;
  const int m0 = blockIdx.x * 16;
  const int n0 = blockIdx.y * 64 + wave * 16;

  v8f acc = {};
#pragma unroll
  for (int k0 = 0; k0 < TDd; k0 += 32) {
    v16h a = load_a_f16(A + (long)m0 * TDd + k0, TDd, lane);
    v16h b = load_b_f32(W + (long)k0 * Cc, Cc, n0, lane);
    acc = wmma_f16(a, b, acc);
  }
  const int hi = lane >> 4;
  const int c  = n0 + (lane & 15);
  const float bc = bo[c];
#pragma unroll
  for (int j = 0; j < 8; ++j) {
    const int r = m0 + j + 8 * hi;
    out[(long)r * Cc + c] = acc[j] + bc;
  }
}

// ---------------------------------------------------------------------------
extern "C" void kernel_launch(void* const* d_in, const int* in_sizes, int n_in,
                              void* d_out, int out_size, void* d_ws,
                              size_t ws_size, hipStream_t stream) {
  const float* q    = (const float*)d_in[0];
  const float* k    = (const float*)d_in[1];
  const float* v    = (const float*)d_in[2];
  const float* pair = (const float*)d_in[3];
  const float* mask = (const float*)d_in[4];
  const float* Wq   = (const float*)d_in[5];
  const float* Wk   = (const float*)d_in[6];
  const float* Wv   = (const float*)d_in[7];
  const float* Wb   = (const float*)d_in[8];
  const float* bbv  = (const float*)d_in[9];
  const float* Wg   = (const float*)d_in[10];
  const float* bg   = (const float*)d_in[11];
  const float* Wo   = (const float*)d_in[12];
  const float* bo   = (const float*)d_in[13];
  float* out = (float*)d_out;

  // workspace layout (bytes)
  char* ws = (char*)d_ws;
  _Float16* bias = (_Float16*)(ws);                       // 2*8*1024*1024*2 = 32 MB
  float*    gate = (float*)(ws + 33554432);               // 2 MB
  _Float16* qp   = (_Float16*)(ws + 35651584);            // 1 MB
  _Float16* kp   = (_Float16*)(ws + 36700160);            // 1 MB
  _Float16* vt   = (_Float16*)(ws + 37748736);            // 1 MB
  _Float16* att  = (_Float16*)(ws + 38797312);            // 1 MB

  // 1) pair bias via WMMA (streams 512 MB once); 16384 wave-tiles, 4/block
  bias_kernel<<<4096, 128, 0, stream>>>(pair, Wb, bbv, bias);

  // 2) projections (WMMA GEMMs)
  dim3 gGemm(128, 4);  // 2048/16 row tiles, 256/64 col tiles
  proj_kernel<<<gGemm, 128, 0, stream>>>(q, Wq, nullptr, qp, nullptr, 0);
  proj_kernel<<<gGemm, 128, 0, stream>>>(k, Wk, nullptr, kp, nullptr, 1);
  proj_kernel<<<gGemm, 128, 0, stream>>>(v, Wv, nullptr, vt, nullptr, 2);
  proj_kernel<<<gGemm, 128, 0, stream>>>(q, Wg, bg, nullptr, gate, 3);

  // 3) fused flash attention (1024 wave-tiles, 4 waves/block)
  attn_kernel<<<256, 128, 0, stream>>>(qp, kp, vt, bias, mask, gate, att);

  // 4) output projection
  out_gemm<<<gGemm, 128, 0, stream>>>(att, Wo, bo, out);
}
